// MatrixLSTMCell_61744449847852
// MI455X (gfx1250) — matrix operand order repaired and checked by reading, not compiled, not run on hardware
//
#include <hip/hip_runtime.h>

typedef __attribute__((ext_vector_type(16))) _Float16 v16h;
typedef __attribute__((ext_vector_type(8)))  _Float16 h8;
typedef __attribute__((ext_vector_type(8)))  float    v8f;

#define BB 4
#define SS 2048
#define DD 512
#define NHH 4
#define DHH 128
#define TDIM 1536   // 3*D

// ---------------- Kernel 1: gate preactivations ----------------
// grid (S, B), 128 threads: wave w computes head w for token (b,s).
__global__ __launch_bounds__(128) void gates_kernel(
    const float* __restrict__ q, const float* __restrict__ k, const float* __restrict__ v,
    const float* __restrict__ Wi, const float* __restrict__ bi,
    const float* __restrict__ Wf, const float* __restrict__ bf,
    float* __restrict__ IPRE, float* __restrict__ FPRE)
{
  const int s = blockIdx.x, b = blockIdx.y;
  const int h = threadIdx.x >> 5, lane = threadIdx.x & 31;
  const float* qs = q + ((long)b * SS + s) * DD;
  const float* ks = k + ((long)b * SS + s) * DD;
  const float* vs = v + ((long)b * SS + s) * DD;
  float si = 0.f, sf = 0.f;
  for (int j = lane; j < TDIM; j += 32) {
    float x = (j < DD) ? qs[j] : (j < 2 * DD) ? ks[j - DD] : vs[j - 2 * DD];
    si += x * Wi[h * TDIM + j];
    sf += x * Wf[h * TDIM + j];
  }
  for (int m = 16; m >= 1; m >>= 1) {
    si += __shfl_xor(si, m, 32);
    sf += __shfl_xor(sf, m, 32);
  }
  if (lane == 0) {
    long idx = ((long)b * NHH + h) * SS + s;
    IPRE[idx] = si + bi[h];
    FPRE[idx] = sf + bf[h];
  }
}

// ---------------- Kernel 2: per-(b,h) scans ----------------
// grid B*NH, 256 threads, 8 elems/thread. Produces a[s]=i-b, amax[t]=runmax(a), m[t]=b+amax.
__global__ __launch_bounds__(256) void scan_kernel(
    const float* __restrict__ IPRE, const float* __restrict__ FPRE,
    float* __restrict__ AARR, float* __restrict__ AMAX, float* __restrict__ MARR)
{
  __shared__ float sh[256];
  const int tid = threadIdx.x;
  const long gb = (long)blockIdx.x * SS;
  const int t0 = tid * 8;
  float cums[8];
  float run = 0.f;
  for (int j = 0; j < 8; ++j) {
    float x = FPRE[gb + t0 + j];
    float ls = fminf(x, 0.f) - log1pf(expf(-fabsf(x)));   // stable log-sigmoid
    run += ls;
    cums[j] = run;
  }
  sh[tid] = run;
  __syncthreads();
  for (int off = 1; off < 256; off <<= 1) {
    float vv = (tid >= off) ? sh[tid - off] : 0.f;
    __syncthreads();
    sh[tid] += vv;
    __syncthreads();
  }
  float excl = sh[tid] - run;
  float av[8], bcv[8], cmax[8];
  float rmax = -3.0e38f;
  for (int j = 0; j < 8; ++j) {
    bcv[j] = excl + cums[j];
    av[j] = IPRE[gb + t0 + j] - bcv[j];
    rmax = fmaxf(rmax, av[j]);
    cmax[j] = rmax;
  }
  __syncthreads();
  sh[tid] = rmax;
  __syncthreads();
  for (int off = 1; off < 256; off <<= 1) {
    float vv = (tid >= off) ? sh[tid - off] : -3.0e38f;
    __syncthreads();
    sh[tid] = fmaxf(sh[tid], vv);
    __syncthreads();
  }
  float exm = (tid > 0) ? sh[tid - 1] : -3.0e38f;
  for (int j = 0; j < 8; ++j) {
    float am = fmaxf(exm, cmax[j]);
    AARR[gb + t0 + j] = av[j];
    AMAX[gb + t0 + j] = am;
    MARR[gb + t0 + j] = bcv[j] + am;
  }
}

// ---------------- Kernel 3: flash-style mLSTM + fused groupnorm ----------------
// grid (S/64, NH, B), 128 threads = 4 waves; wave w owns q-rows [qblk+16w, +16).
__global__ __launch_bounds__(128) void mlstm_kernel(
    const float* __restrict__ Q, const float* __restrict__ K, const float* __restrict__ V,
    const float* __restrict__ AARR, const float* __restrict__ AMAX, const float* __restrict__ MARR,
    const float* __restrict__ LNW, const float* __restrict__ LNB, float* __restrict__ OUT)
{
  __shared__ __attribute__((aligned(32))) _Float16 sQ[64 * DHH];   // [qrow][d]
  __shared__ __attribute__((aligned(32))) _Float16 sK[32 * DHH];   // [key][d]
  __shared__ __attribute__((aligned(32))) _Float16 sVt[DHH * 32];  // [d][key]
  __shared__ __attribute__((aligned(32))) _Float16 sP[4][16 * 32]; // per wave: [qrow][key]

  const int qblk = blockIdx.x * 64;
  const int h = blockIdx.y, b = blockIdx.z;
  const int tid = threadIdx.x, wv = tid >> 5, lane = tid & 31;
  const int ln = lane & 15, lh = lane >> 4;
  const int q0w = qblk + 16 * wv;
  const long gbase = (long)(b * NHH + h) * SS;
  const float scale = 0.08838834764831845f; // 1/sqrt(128)

  for (int i = tid; i < 64 * DHH; i += 128) {
    int r = i >> 7, c = i & 127;
    sQ[i] = (_Float16)Q[((long)b * SS + qblk + r) * DD + h * DHH + c];
  }
  float amaxr[8], mr[8];
  for (int r = 0; r < 8; ++r) {
    int t = q0w + 8 * lh + r;
    amaxr[r] = AMAX[gbase + t];
    mr[r] = MARR[gbase + t];
  }
  __syncthreads();
  // Hoist loop-invariant Q A-fragments (16x128 per wave) into registers.
  v16h qfrag[4];
  for (int dc = 0; dc < 4; ++dc) {
    int ab = (16 * wv + ln) * DHH + 32 * dc + 16 * lh;
    h8 lo = *(const h8*)(&sQ[ab]);
    h8 hi = *(const h8*)(&sQ[ab + 8]);
    qfrag[dc] = __builtin_shufflevector(lo, hi, 0,1,2,3,4,5,6,7,8,9,10,11,12,13,14,15);
  }

  v8f hacc[8] = {};
  v8f rs = {};

  const int nch = qblk / 32 + 2;
  // Per-thread prefetch coordinates: one 128B line per thread covers a 32x128 f32 chunk.
  const int pr = tid >> 2, pc = (tid & 3) * 32;
  for (int ch = 0; ch < nch; ++ch) {
    const int c0 = ch * 32;
    __syncthreads();
    for (int i = tid; i < 32 * DHH; i += 128) {
      int r = i >> 7, c = i & 127;
      long g = ((long)b * SS + c0 + r) * DD + h * DHH + c;
      sK[i] = (_Float16)K[g];
      sVt[c * 32 + r] = (_Float16)V[g];
    }
    __syncthreads();
    // Warm L2/WGP$ with next chunk's K/V lines while this chunk computes.
    if (ch + 1 < nch) {
      long pg = ((long)b * SS + c0 + 32 + pr) * DD + h * DHH + pc;
      __builtin_prefetch(&K[pg], 0, 3);
      __builtin_prefetch(&V[pg], 0, 3);
    }
    if (c0 <= q0w + 15) {
      _Float16* pbuf = &sP[wv][0];
      for (int half = 0; half < 2; ++half) {
        const int k0 = c0 + 16 * half;
        if (k0 <= q0w + 15) {
          float a_s = AARR[gbase + k0 + ln];          // per-column decay term
          // Preload all four 32x16 K B-fragments, then run WMMAs back-to-back.
          v16h bfrag[4];
          for (int dc = 0; dc < 4; ++dc) {
            int bbase = ((k0 - c0) + ln) * DHH + 32 * dc + 16 * lh;
            bfrag[dc] = *(const v16h*)(&sK[bbase]);
          }
          v8f acc = {};
          for (int dc = 0; dc < 4; ++dc)
            acc = __builtin_amdgcn_wmma_f32_16x16x32_f16(false, qfrag[dc], false, bfrag[dc],
                                                         (short)0, acc, false, false);
          const bool diag = (k0 == q0w);
          for (int r = 0; r < 8; ++r) {
            int rr = 8 * lh + r;                       // row within 16-tile
            float val = 0.f;
            if (!diag || (ln <= rr))
              val = acc[r] * scale * __expf(a_s - amaxr[r]);
            rs[r] += val;
            pbuf[rr * 32 + 16 * half + ln] = (_Float16)val;
          }
        } else {
          for (int r = 0; r < 8; ++r)
            pbuf[(8 * lh + r) * 32 + 16 * half + ln] = (_Float16)0.f;
        }
      }
      __asm__ volatile("" ::: "memory");
      __builtin_amdgcn_wave_barrier();   // DS ops are wave-ordered; block compiler reordering
      // P(16x32) @ V(32x128), depth-2 software pipeline on the V fragments so the
      // DS loads for fragment f+1 are in flight while WMMA f executes.
      int pb = ln * 32 + 16 * lh;
      h8 plo = *(const h8*)(&sP[wv][pb]);
      h8 phi = *(const h8*)(&sP[wv][pb + 8]);
      v16h pfrag = __builtin_shufflevector(plo, phi, 0,1,2,3,4,5,6,7,8,9,10,11,12,13,14,15);
      v16h vcur = *(const v16h*)(&sVt[ln * 32 + 16 * lh]);
      for (int f = 0; f < 8; ++f) {
        int fn = (f < 7) ? (f + 1) : 7;
        v16h vnext = *(const v16h*)(&sVt[(16 * fn + ln) * 32 + 16 * lh]);
        hacc[f] = __builtin_amdgcn_wmma_f32_16x16x32_f16(false, pfrag, false, vcur,
                                                         (short)0, hacc[f], false, false);
        vcur = vnext;
      }
    }
  }

  // n = max(|rowsum|, exp(-m)); divide; fused per-head LayerNorm over DH=128
  float invn[8];
  for (int r = 0; r < 8; ++r) {
    float x = rs[r];
    x += __shfl_xor(x, 1, 32); x += __shfl_xor(x, 2, 32);
    x += __shfl_xor(x, 4, 32); x += __shfl_xor(x, 8, 32);
    float n = fmaxf(fabsf(x), __expf(-mr[r]));
    invn[r] = 1.f / (n + 5e-5f);
  }
  for (int f = 0; f < 8; ++f)
    for (int r = 0; r < 8; ++r) hacc[f][r] *= invn[r];

  float mu[8], rstd[8];
  for (int r = 0; r < 8; ++r) {
    float s1 = 0.f, s2 = 0.f;
    for (int f = 0; f < 8; ++f) { float vv = hacc[f][r]; s1 += vv; s2 += vv * vv; }
    s1 += __shfl_xor(s1, 1, 32); s1 += __shfl_xor(s1, 2, 32);
    s1 += __shfl_xor(s1, 4, 32); s1 += __shfl_xor(s1, 8, 32);
    s2 += __shfl_xor(s2, 1, 32); s2 += __shfl_xor(s2, 2, 32);
    s2 += __shfl_xor(s2, 4, 32); s2 += __shfl_xor(s2, 8, 32);
    float m_ = s1 * (1.f / 128.f);
    float var = s2 * (1.f / 128.f) - m_ * m_;
    mu[r] = m_;
    rstd[r] = rsqrtf(var + 1e-3f);
  }
  for (int f = 0; f < 8; ++f) {
    int col = h * DHH + 16 * f + ln;
    float w_ = 1.f + LNW[col];
    float b_ = LNB[col];
    for (int r = 0; r < 8; ++r) {
      int t = q0w + 8 * lh + r;
      OUT[((long)b * SS + t) * DD + col] = (hacc[f][r] - mu[r]) * rstd[r] * w_ + b_;
    }
  }
}

extern "C" void kernel_launch(void* const* d_in, const int* in_sizes, int n_in,
                              void* d_out, int out_size, void* d_ws, size_t ws_size,
                              hipStream_t stream) {
  (void)in_sizes; (void)n_in; (void)out_size; (void)ws_size;
  const float* q   = (const float*)d_in[0];
  const float* k   = (const float*)d_in[1];
  const float* v   = (const float*)d_in[2];
  const float* Wi  = (const float*)d_in[3];
  const float* bi  = (const float*)d_in[4];
  const float* Wf  = (const float*)d_in[5];
  const float* bf  = (const float*)d_in[6];
  const float* lnw = (const float*)d_in[7];
  const float* lnb = (const float*)d_in[8];
  float* out = (float*)d_out;

  const long BNHS = (long)BB * NHH * SS;   // 32768
  float* IPRE = (float*)d_ws;
  float* FPRE = IPRE + BNHS;
  float* AARR = FPRE + BNHS;
  float* AMAX = AARR + BNHS;
  float* MARR = AMAX + BNHS;

  gates_kernel<<<dim3(SS, BB), 128, 0, stream>>>(q, k, v, Wi, bi, Wf, bf, IPRE, FPRE);
  scan_kernel<<<BB * NHH, 256, 0, stream>>>(IPRE, FPRE, AARR, AMAX, MARR);
  mlstm_kernel<<<dim3(SS / 64, NHH, BB), 128, 0, stream>>>(q, k, v, AARR, AMAX, MARR,
                                                           lnw, lnb, out);
}
